// CLIPAttention_50680614093277
// MI455X (gfx1250) — matrix operand order repaired and checked by reading, not compiled, not run on hardware
//
#include <hip/hip_runtime.h>

// ---------------------------------------------------------------------------
// CLIP attention for gfx1250 (MI455X): bf16 WMMA (v_wmma_f32_16x16x32_bf16)
// with fp32 accumulation, flash-style online-softmax attention.
// Round 4: GEMM k-loops use manual depth-1 software pipeline with DEFAULT
// unrolling (compiler 2x-unrolls and rotates double buffers -> partial
// s_wait_loadcnt, no phi copies). Attention keeps K-prefetch + early V loads
// overlapping the softmax VALU block.
// ---------------------------------------------------------------------------

typedef __bf16 bf16;
typedef __attribute__((ext_vector_type(16))) __bf16 v16bf;
typedef __attribute__((ext_vector_type(8)))  __bf16 v8bf;
typedef __attribute__((ext_vector_type(8)))  float  v8f;

#define EMBED 768
#define HEADS 12
#define HDIM  64
#define SEQ   512
#define BATCH 32
#define MROWS (BATCH * SEQ)      // 16384
#define ATTN_SCALE 0.125f        // 64^-0.5

__device__ __forceinline__ bf16 f2bf(float f) {
    union { float f; unsigned u; } a; a.f = f;
    unsigned r = a.u + 0x7FFFu + ((a.u >> 16) & 1u);   // round-to-nearest-even
    unsigned short h = (unsigned short)(r >> 16);
    return __builtin_bit_cast(__bf16, h);
}

union FragU { v16bf v; v8bf h[2]; };

// A-fragment chunks relative to a per-lane base pointer:
//   base = A + (m0 + (lane&15))*lda + ((lane>>4)<<4); chunks at +0 and +8
__device__ __forceinline__ v16bf load_a_at(const bf16* p) {
    FragU f;
    f.h[0] = *(const v8bf*)(p);
    f.h[1] = *(const v8bf*)(p + 8);
    return f.v;
}
// B-fragment ([N][K] storage):
//   base = B + (n0 + (lane&15))*ldb + ((lane>>4)<<4); chunks at +0 and +8
__device__ __forceinline__ v16bf load_b_at(const bf16* p) {
    FragU f;
    f.h[0] = *(const v8bf*)(p);
    f.h[1] = *(const v8bf*)(p + 8);
    return f.v;
}

__device__ __forceinline__ v16bf load_a_frag(const bf16* base, int lda,
                                             int m0, int k0, int lane) {
    return load_a_at(base + (size_t)(m0 + (lane & 15)) * lda + k0 +
                     ((lane >> 4) << 4));
}
__device__ __forceinline__ v16bf load_b_frag(const bf16* base, int ldb,
                                             int n0, int k0, int lane) {
    return load_b_at(base + (size_t)(n0 + (lane & 15)) * ldb + k0 +
                     ((lane >> 4) << 4));
}

__device__ __forceinline__ v8f wmma_bf16(v16bf a, v16bf b, v8f c) {
    return __builtin_amdgcn_wmma_f32_16x16x32_bf16(
        false, a, false, b, (short)0, c, false, false);
}

// ---------------------------------------------------------------------------
// Elementwise conversions
// ---------------------------------------------------------------------------
__global__ void k_f32_to_bf16(const float* __restrict__ in,
                              bf16* __restrict__ out, int n) {
    int i = blockIdx.x * blockDim.x + threadIdx.x;
    if (i < n) out[i] = f2bf(in[i]);
}

// W[k][n] fp32 -> Wt[n][k] bf16   (768x768)
__global__ void k_transpose_w(const float* __restrict__ W,
                              bf16* __restrict__ Wt) {
    int i = blockIdx.x * blockDim.x + threadIdx.x;
    if (i >= EMBED * EMBED) return;
    int k = i / EMBED, n = i % EMBED;
    Wt[(size_t)n * EMBED + k] = f2bf(W[i]);
}

// ---------------------------------------------------------------------------
// Fused QKV projection: 32x16 tile per wave, 3 weights share A-fragments.
// Depth-1 software pipeline; compiler unrolls 2x and rotates buffers.
//   Q -> (acc+bq)*SCALE  stored [b,h,s,d] bf16
//   K -> (acc+bk)        stored [b,h,s,d] bf16
//   V -> (acc+bv)        stored [b,h,d,s] bf16 (transposed for P*V B-frags)
// ---------------------------------------------------------------------------
__global__ __launch_bounds__(32)
void k_qkv_gemm(const bf16* __restrict__ X,
                const bf16* __restrict__ Wqt, const bf16* __restrict__ Wkt,
                const bf16* __restrict__ Wvt,
                const float* __restrict__ bq, const float* __restrict__ bk,
                const float* __restrict__ bv,
                bf16* __restrict__ Qo, bf16* __restrict__ Ko,
                bf16* __restrict__ Vto) {
    int lane = threadIdx.x;
    int m0 = blockIdx.x * 32;       // 0..16352
    int n0 = blockIdx.y * 16;       // 0..752

    const bf16* pa0 = X + (size_t)(m0 + (lane & 15)) * EMBED + ((lane >> 4) << 4);
    const bf16* pa1 = pa0 + (size_t)16 * EMBED;
    size_t boff = (size_t)(n0 + (lane & 15)) * EMBED + ((lane >> 4) << 4);
    const bf16* pq = Wqt + boff;
    const bf16* pk = Wkt + boff;
    const bf16* pv = Wvt + boff;

    v8f cq[2] = {}, ck[2] = {}, cv[2] = {};
    v16bf a0 = load_a_at(pa0), a1 = load_a_at(pa1);
    v16bf wq = load_b_at(pq), wk = load_b_at(pk), wv = load_b_at(pv);

    for (int k0 = 32; k0 < EMBED; k0 += 32) {
        pa0 += 32; pa1 += 32; pq += 32; pk += 32; pv += 32;
        v16bf a0n = load_a_at(pa0), a1n = load_a_at(pa1);
        v16bf wqn = load_b_at(pq), wkn = load_b_at(pk), wvn = load_b_at(pv);
        cq[0] = wmma_bf16(a0, wq, cq[0]);  cq[1] = wmma_bf16(a1, wq, cq[1]);
        ck[0] = wmma_bf16(a0, wk, ck[0]);  ck[1] = wmma_bf16(a1, wk, ck[1]);
        cv[0] = wmma_bf16(a0, wv, cv[0]);  cv[1] = wmma_bf16(a1, wv, cv[1]);
        a0 = a0n; a1 = a1n; wq = wqn; wk = wkn; wv = wvn;
    }
    cq[0] = wmma_bf16(a0, wq, cq[0]);  cq[1] = wmma_bf16(a1, wq, cq[1]);
    ck[0] = wmma_bf16(a0, wk, ck[0]);  ck[1] = wmma_bf16(a1, wk, ck[1]);
    cv[0] = wmma_bf16(a0, wv, cv[0]);  cv[1] = wmma_bf16(a1, wv, cv[1]);

    int col = n0 + (lane & 15);
    int h = col >> 6, d = col & 63;
    float vbq = bq[col], vbk = bk[col], vbv = bv[col];
    int half = (lane >> 4) << 3;
    for (int im = 0; im < 2; ++im) {
        for (int r = 0; r < 8; ++r) {
            int row = m0 + im * 16 + r + half;        // 0..16383
            int bidx = row >> 9, s = row & 511;
            size_t bh = (size_t)bidx * HEADS + h;
            size_t sdoff = (bh * SEQ + s) * HDIM + d;
            Qo[sdoff] = f2bf((cq[im][r] + vbq) * ATTN_SCALE);
            Ko[sdoff] = f2bf(ck[im][r] + vbk);
            Vto[(bh * HDIM + d) * SEQ + s] = f2bf(cv[im][r] + vbv);
        }
    }
}

// ---------------------------------------------------------------------------
// Attention: one wave per (b, h, 16 query rows). Flash-style online softmax.
// j-loop software pipelined: next K tile + current V tile loads issue before
// the softmax VALU block, so VMEM latency hides behind exp/shuffle work.
// Causal mask applied as +(-1e9) above the diagonal (matches reference add).
// Writes context as bf16 [m=b*512+s][e=h*64+d] for the output projection.
// ---------------------------------------------------------------------------
__global__ __launch_bounds__(32)
void k_attention(const bf16* __restrict__ Q, const bf16* __restrict__ K,
                 const bf16* __restrict__ Vt, bf16* __restrict__ Ctx) {
    __shared__ bf16 pbuf[16 * 32];   // P tile: C-layout -> A-layout transpose
    int lane = threadIdx.x;
    int qt = blockIdx.x;             // 0..31
    int h  = blockIdx.y;             // 0..11
    int b  = blockIdx.z;             // 0..31
    size_t bh = (size_t)b * HEADS + h;
    const bf16* q  = Q  + bh * SEQ * HDIM;
    const bf16* kk = K  + bh * SEQ * HDIM;
    const bf16* vt = Vt + bh * HDIM * SEQ;
    int q0 = qt * 16;

    // Q A-fragments for the full head dim (K = 0..31 and 32..63)
    v16bf qa0 = load_a_frag(q, HDIM, q0, 0,  lane);
    v16bf qa1 = load_a_frag(q, HDIM, q0, 32, lane);

    float m[8], l[8];
    v8f o[4] = {};
    for (int r = 0; r < 8; ++r) { m[r] = -3.0e38f; l[r] = 0.0f; }

    int half  = (lane >> 4) << 3;                 // row offset per half
    int colid = lane & 15;
    int ntiles = (q0 + 47) / 32;                  // causal: skip fully-masked

    // prime the K-tile pipeline
    v16bf kb0 = load_b_frag(kk, HDIM, 0, 0, lane);
    v16bf kb1 = load_b_frag(kk, HDIM, 0, 32, lane);
    v16bf kb2 = load_b_frag(kk, HDIM, 16, 0, lane);
    v16bf kb3 = load_b_frag(kk, HDIM, 16, 32, lane);

    for (int t = 0; t < ntiles; ++t) {
        int jb = t * 32;
        v8f slo = {}, shi = {};
        slo = wmma_bf16(qa0, kb0, slo);
        slo = wmma_bf16(qa1, kb1, slo);
        shi = wmma_bf16(qa0, kb2, shi);
        shi = wmma_bf16(qa1, kb3, shi);

        // V fragments for this tile: independent of softmax, issue early
        v16bf vb0 = load_b_frag(vt, SEQ, 0,  jb, lane);
        v16bf vb1 = load_b_frag(vt, SEQ, 16, jb, lane);
        v16bf vb2 = load_b_frag(vt, SEQ, 32, jb, lane);
        v16bf vb3 = load_b_frag(vt, SEQ, 48, jb, lane);
        // prefetch next K tile behind the softmax VALU block
        if (t + 1 < ntiles) {
            int jn = jb + 32;
            kb0 = load_b_frag(kk, HDIM, jn, 0, lane);
            kb1 = load_b_frag(kk, HDIM, jn, 32, lane);
            kb2 = load_b_frag(kk, HDIM, jn + 16, 0, lane);
            kb3 = load_b_frag(kk, HDIM, jn + 16, 32, lane);
        }

        int colL = jb + colid, colH = colL + 16;
        for (int r = 0; r < 8; ++r) {
            int row = q0 + r + half;
            float sl = slo[r] + ((colL > row) ? -1.0e9f : 0.0f);
            float sh = shi[r] + ((colH > row) ? -1.0e9f : 0.0f);
            // row max across the 16 lanes of this half
            float mx = fmaxf(sl, sh);
            for (int off = 1; off < 16; off <<= 1)
                mx = fmaxf(mx, __shfl_xor(mx, off, 32));
            float mnew = fmaxf(m[r], mx);
            float alpha = __expf(m[r] - mnew);
            float pl = __expf(sl - mnew);
            float ph = __expf(sh - mnew);
            float rs = pl + ph;
            for (int off = 1; off < 16; off <<= 1)
                rs += __shfl_xor(rs, off, 32);
            l[r] = l[r] * alpha + rs;
            m[r] = mnew;
            for (int dt = 0; dt < 4; ++dt) o[dt][r] *= alpha;
            int rl = r + half;
            pbuf[rl * 32 + colid]      = f2bf(pl);
            pbuf[rl * 32 + colid + 16] = f2bf(ph);
        }
        __syncthreads();
        v16bf pa = load_a_frag(pbuf, 32, 0, 0, lane);
        __syncthreads();
        o[0] = wmma_bf16(pa, vb0, o[0]);
        o[1] = wmma_bf16(pa, vb1, o[1]);
        o[2] = wmma_bf16(pa, vb2, o[2]);
        o[3] = wmma_bf16(pa, vb3, o[3]);
    }

    for (int r = 0; r < 8; ++r) {
        float inv = 1.0f / l[r];
        int row = q0 + r + half;                      // s within batch b
        size_t mrow = (size_t)b * SEQ + row;
        for (int dt = 0; dt < 4; ++dt) {
            int e = h * HDIM + dt * 16 + colid;
            Ctx[mrow * EMBED + e] = f2bf(o[dt][r] * inv);
        }
    }
}

// ---------------------------------------------------------------------------
// Output projection: Ctx[16384x768] * Wo^T + bo -> fp32 out
// 32x32 tile per wave, depth-1 software pipeline (compiler-rotated).
// ---------------------------------------------------------------------------
__global__ __launch_bounds__(32)
void k_out_gemm(const bf16* __restrict__ Ctx, const bf16* __restrict__ Wot,
                const float* __restrict__ bo, float* __restrict__ Out) {
    int lane = threadIdx.x;
    int m0 = blockIdx.x * 32;
    int n0 = blockIdx.y * 32;

    const bf16* pa0 = Ctx + (size_t)(m0 + (lane & 15)) * EMBED + ((lane >> 4) << 4);
    const bf16* pa1 = pa0 + (size_t)16 * EMBED;
    const bf16* pb0 = Wot + (size_t)(n0 + (lane & 15)) * EMBED + ((lane >> 4) << 4);
    const bf16* pb1 = pb0 + (size_t)16 * EMBED;

    v8f c00 = {}, c01 = {}, c10 = {}, c11 = {};
    v16bf a0 = load_a_at(pa0), a1 = load_a_at(pa1);
    v16bf b0 = load_b_at(pb0), b1 = load_b_at(pb1);

    for (int k0 = 32; k0 < EMBED; k0 += 32) {
        pa0 += 32; pa1 += 32; pb0 += 32; pb1 += 32;
        v16bf a0n = load_a_at(pa0), a1n = load_a_at(pa1);
        v16bf b0n = load_b_at(pb0), b1n = load_b_at(pb1);
        c00 = wmma_bf16(a0, b0, c00);  c01 = wmma_bf16(a0, b1, c01);
        c10 = wmma_bf16(a1, b0, c10);  c11 = wmma_bf16(a1, b1, c11);
        a0 = a0n; a1 = a1n; b0 = b0n; b1 = b1n;
    }
    c00 = wmma_bf16(a0, b0, c00);  c01 = wmma_bf16(a0, b1, c01);
    c10 = wmma_bf16(a1, b0, c10);  c11 = wmma_bf16(a1, b1, c11);

    int colL = n0 + (lane & 15);
    float vb0 = bo[colL], vb1 = bo[colL + 16];
    int half = (lane >> 4) << 3;
    for (int r = 0; r < 8; ++r) {
        int row0 = m0 + r + half;
        int row1 = row0 + 16;
        Out[(size_t)row0 * EMBED + colL]      = c00[r] + vb0;
        Out[(size_t)row0 * EMBED + colL + 16] = c01[r] + vb1;
        Out[(size_t)row1 * EMBED + colL]      = c10[r] + vb0;
        Out[(size_t)row1 * EMBED + colL + 16] = c11[r] + vb1;
    }
}

// ---------------------------------------------------------------------------
// Host launch
// ---------------------------------------------------------------------------
extern "C" void kernel_launch(void* const* d_in, const int* in_sizes, int n_in,
                              void* d_out, int out_size, void* d_ws, size_t ws_size,
                              hipStream_t stream) {
    const float* hs = (const float*)d_in[0];
    // d_in[1] = causal mask (applied analytically in-kernel)
    const float* Wq = (const float*)d_in[2];
    const float* bq = (const float*)d_in[3];
    const float* Wk = (const float*)d_in[4];
    const float* bk = (const float*)d_in[5];
    const float* Wv = (const float*)d_in[6];
    const float* bv = (const float*)d_in[7];
    const float* Wo = (const float*)d_in[8];
    const float* bo = (const float*)d_in[9];
    float* out = (float*)d_out;

    char* ws = (char*)d_ws;
    size_t off = 0;
    const size_t XB  = (size_t)MROWS * EMBED * sizeof(bf16);   // 25.2 MB
    const size_t WB  = (size_t)EMBED * EMBED * sizeof(bf16);   // 1.18 MB
    bf16* Xb  = (bf16*)(ws + off); off += XB;
    bf16* Wqt = (bf16*)(ws + off); off += WB;
    bf16* Wkt = (bf16*)(ws + off); off += WB;
    bf16* Wvt = (bf16*)(ws + off); off += WB;
    bf16* Wot = (bf16*)(ws + off); off += WB;
    bf16* Qw  = (bf16*)(ws + off); off += XB;
    bf16* Kw  = (bf16*)(ws + off); off += XB;
    bf16* Vtw = (bf16*)(ws + off); off += XB;
    bf16* Cx  = (bf16*)(ws + off); off += XB;
    (void)ws_size; (void)n_in; (void)in_sizes; (void)out_size;

    // 1) convert inputs to bf16 (weights transposed to [N][K])
    {
        int n = MROWS * EMBED;
        k_f32_to_bf16<<<(n + 255) / 256, 256, 0, stream>>>(hs, Xb, n);
        int wn = EMBED * EMBED;
        int wg = (wn + 255) / 256;
        k_transpose_w<<<wg, 256, 0, stream>>>(Wq, Wqt);
        k_transpose_w<<<wg, 256, 0, stream>>>(Wk, Wkt);
        k_transpose_w<<<wg, 256, 0, stream>>>(Wv, Wvt);
        k_transpose_w<<<wg, 256, 0, stream>>>(Wo, Wot);
    }
    // 2) fused QKV projection (32x16 tiles, pipelined)
    k_qkv_gemm<<<dim3(MROWS / 32, EMBED / 16), 32, 0, stream>>>(
        Xb, Wqt, Wkt, Wvt, bq, bk, bv, Qw, Kw, Vtw);
    // 3) attention (flash-style, causal, pipelined)
    k_attention<<<dim3(SEQ / 16, HEADS, BATCH), 32, 0, stream>>>(Qw, Kw, Vtw, Cx);
    // 4) output projection (32x32 tiles, pipelined)
    k_out_gemm<<<dim3(MROWS / 32, EMBED / 32), 32, 0, stream>>>(Cx, Wot, bo, out);
}
